// AMPBlock1_81655918231624
// MI455X (gfx1250) — matrix operand order repaired and checked by reading, not compiled, not run on hardware
//
#include <hip/hip_runtime.h>
#include <hip/hip_bf16.h>
#include <math.h>

// ---------------- problem constants ----------------
#define CC   512      // channels
#define TT   8192     // time
#define BB   8        // batch
#define FK   12       // filter taps

typedef __attribute__((ext_vector_type(16))) __bf16 v16bf;
typedef __attribute__((ext_vector_type(8)))  float  v8f;
typedef __attribute__((ext_vector_type(4)))  int    v4i;
typedef __attribute__((address_space(1))) v4i  g_v4i;   // global v4i
typedef __attribute__((address_space(3))) v4i  l_v4i;   // LDS v4i
typedef __attribute__((address_space(3))) void lds_void;

struct Filt { float up[FK]; float dn[FK]; };

struct alignas(16) U4 { unsigned int x, y, z, w; };
union FragBF { v16bf v; U4 q[2]; };

static __device__ __forceinline__ unsigned short f2bf(float f) {
    unsigned int u = __float_as_uint(f);
    unsigned int r = u + 0x7FFFu + ((u >> 16) & 1u);   // round to nearest even
    return (unsigned short)(r >> 16);
}

static __device__ __forceinline__ v8f wmma_bf16(v16bf a, v16bf b, v8f c) {
    return __builtin_amdgcn_wmma_f32_16x16x32_bf16(
        false, a, false, b, (short)0, c, false, false);
}

// ---- CDNA5 async global->LDS copy (16B per lane), ASYNCcnt-tracked ----
static __device__ __forceinline__ void async_cp16(unsigned short* l,
                                                  const unsigned short* g) {
#if __has_builtin(__builtin_amdgcn_global_load_async_to_lds_b128)
    __builtin_amdgcn_global_load_async_to_lds_b128(
        (g_v4i*)(uintptr_t)g,            // AS1, non-const, v4i-typed
        (l_v4i*)(lds_void*)l,            // AS3, v4i-typed
        0, 0);
#else
    asm volatile("global_load_async_to_lds_b128 %0, %1, off"
                 :
                 : "v"((lds_void*)l), "v"(g)
                 : "memory");
#endif
}

static __device__ __forceinline__ void wait_async0() {
#if __has_builtin(__builtin_amdgcn_s_wait_asynccnt)
    __builtin_amdgcn_s_wait_asynccnt(0);
#else
    asm volatile("s_wait_asynccnt 0x0" ::: "memory");
#endif
}

// =====================================================================
// Fused activation1d: up x2 (12-tap, edge pad) -> snakebeta -> down x2.
// Emits bf16 directly (conv input precision) to halve traffic.
// =====================================================================
__global__ void __launch_bounds__(256)
act1d_kernel(const float* __restrict__ X,
             const float* __restrict__ alpha,
             const float* __restrict__ beta,
             unsigned short* __restrict__ Y, Filt f)
{
    __shared__ float sx[272];
    const int t0 = blockIdx.x * 256;
    const int c  = blockIdx.y;
    const int b  = blockIdx.z;
    const int tid = threadIdx.x;

    const size_t rowbase = ((size_t)(b * CC + c)) * TT;
    for (int jj = tid; jj < 272; jj += 256) {
        int gi = t0 - 5 + jj;
        gi = gi < 0 ? 0 : (gi > TT - 1 ? TT - 1 : gi);
        sx[jj] = X[rowbase + gi];
    }
    __syncthreads();

    const float a    = __expf(alpha[c]);
    const float binv = 1.0f / (__expf(beta[c]) + 1e-9f);

    const int t = t0 + tid;
    float acc = 0.0f;
#pragma unroll
    for (int m = 0; m < FK; ++m) {
        int s = 2 * t - 5 + m;
        s = s < 0 ? 0 : (s > 2 * TT - 1 ? 2 * TT - 1 : s);
        const int p = s & 1;
        float u = 0.0f;
#pragma unroll
        for (int kk = 0; kk < 6; ++kk) {
            const int k = p + 2 * kk;
            int iloc = ((s + 4 + k) >> 1) - t0;     // == i - (t0-5)
            iloc = iloc < 0 ? 0 : (iloc > 271 ? 271 : iloc);
            u = fmaf(f.up[k], sx[iloc], u);
        }
        const float sn = __sinf(u * a);
        const float v  = u + sn * sn * binv;
        acc = fmaf(f.dn[m], v, acc);
    }
    Y[rowbase + t] = f2bf(acc);
}

// =====================================================================
// Pack fp32 weights [C][C][3] -> bf16 tiles laid out exactly as the conv
// kernel's LDS wants them: [cb][cc][k][co(32)][ci(32)] (linear per tile).
// =====================================================================
__global__ void __launch_bounds__(256)
wpack_kernel(const float* __restrict__ w, unsigned short* __restrict__ wp)
{
    const int o  = blockIdx.x * 256 + threadIdx.x;    // 786432 total
    const int tl = o / 3072;
    const int r  = o - tl * 3072;
    const int k  = r >> 10;
    const int rr = r & 1023;
    const int co = rr >> 5;
    const int ci = rr & 31;
    const int cb = tl >> 4;
    const int cc = tl & 15;
    const float v = w[(((size_t)(cb * 32 + co)) * CC + (cc * 32 + ci)) * 3 + k];
    wp[o] = f2bf(v);
}

// =====================================================================
// Dilated conv1d (K=3) as bf16 WMMA GEMM, f32 accumulate.
// WG: 8 waves = 32co x 128t; wave = 16co x 32t; 96 wmma/wave.
// Weights: async global->LDS (pre-packed bf16 tiles).
// Input: bf16, coalesced 16B loads + LDS store-transpose [t][ci].
// =====================================================================
__global__ void __launch_bounds__(256)
conv1d_wmma_kernel(const unsigned short* __restrict__ in,   // bf16 [B][C][T]
                   const unsigned short* __restrict__ wp,   // packed bf16
                   const float* __restrict__ bias,
                   const float* __restrict__ res,
                   float* __restrict__ out,
                   int dil, int has_res)
{
    __shared__ __align__(16) unsigned short xs[160 * 32];    // [trow][ci] bf16
    __shared__ __align__(16) unsigned short wl[3 * 32 * 32]; // [k][co][ci] bf16

    const int tid  = threadIdx.x;
    const int lane = tid & 31;
    const int wave = tid >> 5;
    const int half = lane >> 4;
    const int ln15 = lane & 15;

    const int wCo = (wave >> 2) * 16;
    const int wT  = (wave & 3) * 32;

    const int tg0 = blockIdx.x * 128;
    const int co0 = blockIdx.y * 32;
    const int cb  = blockIdx.y;
    const int b   = blockIdx.z;
    const int t_start = tg0 - 8;         // 16B-aligned window base, covers halo

    v8f acc0 = {};
    v8f acc1 = {};

    for (int cc = 0; cc < CC / 32; ++cc) {
        // ---- weights: pure linear async copy, 6144B (384 x 16B lanes) ----
        {
            const unsigned short* src = wp + ((size_t)(cb * 16 + cc)) * 3072;
            for (int task = tid; task < 384; task += 256)
                async_cp16(&wl[task * 8], src + task * 8);
        }
        // ---- input: coalesced bf16 16B chunks -> transpose into LDS ----
        for (int task = tid; task < 640; task += 256) {  // 32 ci x 20 chunks
            const int ci    = task / 20;
            const int chunk = task - ci * 20;
            const int tbase = t_start + chunk * 8;
            const size_t gb = ((size_t)(b * CC + cc * 32 + ci)) * TT;
            unsigned short e[8];
            if (tbase >= 0 && tbase + 7 < TT) {
                U4 u = *reinterpret_cast<const U4*>(in + gb + tbase);
                e[0] = (unsigned short)(u.x);       e[1] = (unsigned short)(u.x >> 16);
                e[2] = (unsigned short)(u.y);       e[3] = (unsigned short)(u.y >> 16);
                e[4] = (unsigned short)(u.z);       e[5] = (unsigned short)(u.z >> 16);
                e[6] = (unsigned short)(u.w);       e[7] = (unsigned short)(u.w >> 16);
            } else {
#pragma unroll
                for (int j = 0; j < 8; ++j) {
                    const int t = tbase + j;
                    e[j] = (t >= 0 && t < TT) ? in[gb + t] : (unsigned short)0;
                }
            }
            const int row0 = chunk * 8;
#pragma unroll
            for (int j = 0; j < 8; ++j)
                xs[(row0 + j) * 32 + ci] = e[j];
        }
        wait_async0();
        __syncthreads();

#pragma unroll
        for (int k = 0; k < 3; ++k) {
            FragBF A;
            const unsigned short* ap = &wl[(k * 32 + wCo + ln15) * 32];
            A.q[0] = *reinterpret_cast<const U4*>(ap + 16 * half);
            A.q[1] = *reinterpret_cast<const U4*>(ap + 16 * half + 8);

            const int trow0 = wT + ln15 + (k - 1) * dil + 8;   // window-relative
            FragBF B0, B1;
            const unsigned short* bp0 = &xs[trow0 * 32 + 16 * half];
            const unsigned short* bp1 = &xs[(trow0 + 16) * 32 + 16 * half];
            B0.q[0] = *reinterpret_cast<const U4*>(bp0);
            B0.q[1] = *reinterpret_cast<const U4*>(bp0 + 8);
            B1.q[0] = *reinterpret_cast<const U4*>(bp1);
            B1.q[1] = *reinterpret_cast<const U4*>(bp1 + 8);

            acc0 = wmma_bf16(A.v, B0.v, acc0);
            acc1 = wmma_bf16(A.v, B1.v, acc1);
        }
        __syncthreads();
    }

    // ---- epilogue: bias (+ residual) + store ----
    const int tcol = tg0 + wT + ln15;
#pragma unroll
    for (int r = 0; r < 8; ++r) {
        const int co = co0 + wCo + r + 8 * half;
        const float bv = bias[co];
        const size_t base = ((size_t)(b * CC + co)) * TT;
        float v0 = acc0[r] + bv;
        float v1 = acc1[r] + bv;
        if (has_res) {
            v0 += res[base + tcol];
            v1 += res[base + tcol + 16];
        }
        out[base + tcol]      = v0;
        out[base + tcol + 16] = v1;
    }
}

// =====================================================================
// Host side
// =====================================================================
static double bessel_i0(double x) {
    double s = 1.0, t = 1.0;
    for (int k = 1; k < 32; ++k) { t *= (x * x) / (4.0 * (double)k * (double)k); s += t; }
    return s;
}

static void make_kaiser_sinc(float* f) {   // cutoff=0.25, half_width=0.3, K=12
    const double cutoff = 0.25, half_width = 0.3;
    const int K = FK, half_size = FK / 2;
    const double delta_f = 4.0 * half_width;
    const double A = 2.285 * (half_size - 1) * M_PI * delta_f + 7.95;
    double beta;
    if (A > 50.0)      beta = 0.1102 * (A - 8.7);
    else if (A >= 21.) beta = 0.5842 * pow(A - 21.0, 0.4) + 0.07886 * (A - 21.0);
    else               beta = 0.0;
    const double i0b = bessel_i0(beta);
    double tmp[FK], sum = 0.0;
    for (int n = 0; n < K; ++n) {
        double r = 2.0 * n / (K - 1) - 1.0;
        double arg = 1.0 - r * r; if (arg < 0) arg = 0;
        double wnd = bessel_i0(beta * sqrt(arg)) / i0b;
        double t = (n - half_size) + 0.5;
        double xx = 2.0 * cutoff * t;
        double sc = (xx == 0.0) ? 1.0 : sin(M_PI * xx) / (M_PI * xx);
        tmp[n] = 2.0 * cutoff * wnd * sc;
        sum += tmp[n];
    }
    for (int n = 0; n < K; ++n) f[n] = (float)(tmp[n] / sum);
}

extern "C" void kernel_launch(void* const* d_in, const int* in_sizes, int n_in,
                              void* d_out, int out_size, void* d_ws, size_t ws_size,
                              hipStream_t stream)
{
    (void)in_sizes; (void)n_in; (void)out_size; (void)ws_size;

    Filt flt;
    float base[FK];
    make_kaiser_sinc(base);
    for (int i = 0; i < FK; ++i) { flt.up[i] = base[i] * 2.0f; flt.dn[i] = base[i]; }

    const float* x = (const float*)d_in[0];
    float* out = (float*)d_out;

    const size_t N  = (size_t)BB * CC * TT;       // 33.5M elements
    const size_t NW = (size_t)CC * CC * 3;        // 786432 weights per conv
    unsigned short* W0 = (unsigned short*)d_ws;                 // bf16 act out
    float*          W1 = (float*)((char*)d_ws + N * 2);         // conv1 out f32
    unsigned short* WP = (unsigned short*)((char*)W1 + N * 4);  // 6 packed w

    const dim3 actGrid(TT / 256, CC, BB), blk(256);
    const dim3 cvGrid(TT / 128, CC / 32, BB);
    const dim3 wpGrid((unsigned)(NW / 256));

    // pre-pack all 6 weight tensors to bf16 WMMA-tile order
    for (int li = 0; li < 3; ++li) {
        wpack_kernel<<<wpGrid, blk, 0, stream>>>((const float*)d_in[1 + 4 * li],
                                                 WP + (2 * li) * NW);
        wpack_kernel<<<wpGrid, blk, 0, stream>>>((const float*)d_in[3 + 4 * li],
                                                 WP + (2 * li + 1) * NW);
    }

    const int dils[3] = {1, 3, 5};
    const float* cur = x;
    for (int li = 0; li < 3; ++li) {
        const float* b1 = (const float*)d_in[2 + 4 * li];
        const float* b2 = (const float*)d_in[4 + 4 * li];
        const float* aA = (const float*)d_in[13 + 4 * li];
        const float* bA = (const float*)d_in[14 + 4 * li];
        const float* aB = (const float*)d_in[15 + 4 * li];
        const float* bB = (const float*)d_in[16 + 4 * li];

        act1d_kernel<<<actGrid, blk, 0, stream>>>(cur, aA, bA, W0, flt);
        conv1d_wmma_kernel<<<cvGrid, blk, 0, stream>>>(W0, WP + (2 * li) * NW,
                                                       b1, nullptr, W1,
                                                       dils[li], 0);
        act1d_kernel<<<actGrid, blk, 0, stream>>>(W1, aB, bB, W0, flt);
        // out = conv(W0) + cur  (element-exact residual -> in-place safe)
        conv1d_wmma_kernel<<<cvGrid, blk, 0, stream>>>(W0, WP + (2 * li + 1) * NW,
                                                       b2, cur, out, 1, 1);
        cur = out;
    }
}
